// EpisodicMemory_81484119540418
// MI455X (gfx1250) — matrix operand order repaired and checked by reading, not compiled, not run on hardware
//
#include <hip/hip_runtime.h>

// ---------------------------------------------------------------------------
// EpisodicMemory forward for MI455X (gfx1250): bf16 WMMA (f32 accum) GEMMs,
// wave32 shuffle reductions, LDS-resident fused attention tiles.
// ---------------------------------------------------------------------------

typedef __bf16 bf16;
typedef __attribute__((ext_vector_type(16))) __bf16 v16bf;
typedef __attribute__((ext_vector_type(8)))  __bf16 v8bf;
typedef __attribute__((ext_vector_type(8)))  float  v8f;

#define BSB 16
#define NCC 2048
#define DD  128
#define MM  1024
#define CC2 64
#define NEGV (-1.0e9f)

__device__ __forceinline__ v8f wmma_bf(v16bf a, v16bf b, v8f c) {
  return __builtin_amdgcn_wmma_f32_16x16x32_bf16(false, a, false, b, (short)0, c,
                                                 false, false);
}

// A fragment (16x32 bf16, MxK): lane<16 holds row=lane, K = k0+{0..7,16..23};
// lane>=16 holds row=lane-16, K = k0+{8..15,24..31}.  rowp = base of this
// lane's row (caller adds row*ld).
__device__ __forceinline__ v16bf frag_a(const bf16* rowp, int k0, int lane) {
  const int h = (lane & 16) ? 8 : 0;
  v8bf lo = *(const v8bf*)(rowp + k0 + h);
  v8bf hi = *(const v8bf*)(rowp + k0 + 16 + h);
  v16bf r;
#pragma unroll
  for (int i = 0; i < 8; ++i) { r[i] = lo[i]; r[i + 8] = hi[i]; }
  return r;
}

// B fragment (32x16 bf16, KxN): lane<16 = col lane, K = k0+{0..7,16..23};
// lane>=16 = col lane-16, K = k0+{8..15,24..31}.  colp = base of this lane's
// column (col-major, contiguous in K) -> two 16B loads.
__device__ __forceinline__ v16bf frag_b(const bf16* colp, int k0, int lane) {
  const int h = (lane & 16) ? 8 : 0;
  v8bf lo = *(const v8bf*)(colp + k0 + h);
  v8bf hi = *(const v8bf*)(colp + k0 + 16 + h);
  v16bf r;
#pragma unroll
  for (int i = 0; i < 8; ++i) { r[i] = lo[i]; r[i + 8] = hi[i]; }
  return r;
}

__device__ __forceinline__ float wave_max32(float x) {
#pragma unroll
  for (int off = 1; off < 32; off <<= 1) x = fmaxf(x, __shfl_xor(x, off, 32));
  return x;
}
__device__ __forceinline__ float wave_sum32(float x) {
#pragma unroll
  for (int off = 1; off < 32; off <<= 1) x += __shfl_xor(x, off, 32);
  return x;
}

// ---------------------------------------------------------------------------
// K0a: f32 -> bf16 convert
__global__ void cvt_bf16_kernel(bf16* __restrict__ dst, const float* __restrict__ src, int n) {
  int i = blockIdx.x * blockDim.x + threadIdx.x;
  if (i < n) dst[i] = (bf16)src[i];
}

// K0b: batched transpose + convert: dst[b][c][r] = src[b][r][c]
__global__ void tcvt_kernel(bf16* __restrict__ dst, const float* __restrict__ src,
                            int rows, int cols) {
  int b = blockIdx.y;
  int i = blockIdx.x * blockDim.x + threadIdx.x;
  if (i >= rows * cols) return;
  int r = i / cols, c = i % cols;
  size_t base = (size_t)b * rows * cols;
  dst[base + (size_t)c * rows + r] = (bf16)src[base + i];
}

// ---------------------------------------------------------------------------
// K1: qsum = q + q@Wq[b%4] + bq   (stored bf16; logits GEMM input)
__global__ __launch_bounds__(128) void qcross_kernel(
    const float* __restrict__ q, const float* __restrict__ bq,
    const bf16* __restrict__ q_bf, const bf16* __restrict__ Wqt_bf,
    bf16* __restrict__ qsum_bf) {
  int b = blockIdx.y, nt = blockIdx.x;
  int lane = threadIdx.x & 31, wave = threadIdx.x >> 5;
  int cblk = b & 3;
  int mh = (lane & 16) ? 8 : 0;
  const bf16* qrow = q_bf + ((size_t)b * NCC + nt * 16 + (lane & 15)) * DD;
  v16bf aq[4];
#pragma unroll
  for (int kk = 0; kk < 4; ++kk) aq[kk] = frag_a(qrow, kk * 32, lane);
  for (int t = wave; t < 8; t += 4) {
    int e = t * 16 + (lane & 15);
    const bf16* wcol = Wqt_bf + ((size_t)cblk * DD + e) * DD;
    v8f acc = {};
#pragma unroll
    for (int kk = 0; kk < 4; ++kk)
      acc = wmma_bf(aq[kk], frag_b(wcol, kk * 32, lane), acc);
    float bias = bq[cblk * DD + e];
#pragma unroll
    for (int v = 0; v < 8; ++v) {
      int r = mh + v;
      size_t idx = ((size_t)b * NCC + nt * 16 + r) * DD + e;
      qsum_bf[idx] = (bf16)(q[idx] + acc[v] + bias);
    }
  }
}

// ---------------------------------------------------------------------------
// K2: fused attention read path + novelty scoring, one 16-row tile per block.
__global__ __launch_bounds__(256) void attn_kernel(
    const float* __restrict__ em_S, const float* __restrict__ surprise,
    const float* __restrict__ w_nov, const float* __restrict__ bo,
    const bf16* __restrict__ q_bf, const bf16* __restrict__ qsum_bf,
    const bf16* __restrict__ qnov_bf, const bf16* __restrict__ emK_bf,
    const bf16* __restrict__ emVt_bf, const bf16* __restrict__ Wot_bf,
    float* __restrict__ out, float* __restrict__ novelty) {
  extern __shared__ __align__(16) unsigned char smem[];
  float* s_sc  = (float*)smem;        // 16*1024 scores (later: bf16 attn)
  float* s_lg  = s_sc + 16 * MM;      // 16*1024 logits/p (later: bf16 out tile)
  float* s_thr = s_lg + 16 * MM;      // 16
  float* s_red = s_thr + 16;          // 16*8

  int b = blockIdx.y, nt = blockIdx.x;
  int lane = threadIdx.x & 31, wave = threadIdx.x >> 5;
  int cblk = b & 3;
  int row0 = nt * 16;
  int mh = (lane & 16) ? 8 : 0;

  // ---- phase 1: scores / logits / sim GEMMs (shared B fragments) ----
  const bf16* qrow  = q_bf    + ((size_t)b * NCC + row0 + (lane & 15)) * DD;
  const bf16* qsrow = qsum_bf + ((size_t)b * NCC + row0 + (lane & 15)) * DD;
  const bf16* qnrow = qnov_bf + ((size_t)b * NCC + row0 + (lane & 15)) * DD;
  v16bf aq[4], am[4], an[4];
#pragma unroll
  for (int kk = 0; kk < 4; ++kk) {
    aq[kk] = frag_a(qrow , kk * 32, lane);
    am[kk] = frag_a(qsrow, kk * 32, lane);
    an[kk] = frag_a(qnrow, kk * 32, lane);
  }
  float rmax[8];
#pragma unroll
  for (int v = 0; v < 8; ++v) rmax[v] = -1.0f;
  for (int t = wave; t < MM / 16; t += 8) {
    int slot = t * 16 + (lane & 15);
    const bf16* kcol = emK_bf + ((size_t)b * MM + slot) * DD;
    v8f accS = {}, accL = {}, accV = {};
#pragma unroll
    for (int kk = 0; kk < 4; ++kk) {
      v16bf bk = frag_b(kcol, kk * 32, lane);
      accS = wmma_bf(aq[kk], bk, accS);
      accL = wmma_bf(am[kk], bk, accL);
      accV = wmma_bf(an[kk], bk, accV);
    }
    bool act = em_S[(size_t)b * MM + slot] > 0.0f;
#pragma unroll
    for (int v = 0; v < 8; ++v) {
      int r = mh + v;
      s_sc[r * MM + slot] = act ? accS[v] : NEGV;
      s_lg[r * MM + slot] = act ? accL[v] : NEGV;
      rmax[v] = fmaxf(rmax[v], act ? accV[v] : -1.0f);
    }
  }
#pragma unroll
  for (int v = 0; v < 8; ++v) {
    float x = rmax[v];
#pragma unroll
    for (int off = 1; off < 16; off <<= 1) x = fmaxf(x, __shfl_xor(x, off, 32));
    rmax[v] = x;
  }
  if ((lane & 15) == 0) {
#pragma unroll
    for (int v = 0; v < 8; ++v) s_red[(mh + v) * 8 + wave] = rmax[v];
  }
  __syncthreads();

  // ---- phase 2a: novelty = w*surprise + (1-w)*(1-clip(max_sim,0)) ----
  if (threadIdx.x < 16) {
    int r = threadIdx.x;
    float mx = -1.0f;
    for (int w2 = 0; w2 < 8; ++w2) mx = fmaxf(mx, s_red[r * 8 + w2]);
    float ms = fmaxf(mx, 0.0f);
    size_t gi = (size_t)b * NCC + row0 + r;
    float wn = w_nov[gi];
    novelty[gi] = wn * surprise[gi] + (1.0f - wn) * (1.0f - ms);
  }

  // ---- phase 2b: per-row 32nd-largest score threshold ----
  for (int rr = 0; rr < 2; ++rr) {
    int r = wave * 2 + rr;
    float vals[32];
#pragma unroll
    for (int i = 0; i < 32; ++i) vals[i] = s_sc[r * MM + lane + i * 32];
    float cur = 3.0e38f;
    for (int it = 0; it < 32; ++it) {
      float m = -3.0e38f;
#pragma unroll
      for (int i = 0; i < 32; ++i) {
        float x = vals[i];
        if (x < cur && x > m) m = x;
      }
      cur = wave_max32(m);
    }
    if (lane == 0) s_thr[r] = cur;
  }
  __syncthreads();

  // ---- phase 3: masked softmax, normalized p kept f32 in s_lg ----
  for (int rr = 0; rr < 2; ++rr) {
    int r = wave * 2 + rr;
    float thr = s_thr[r];
    float mx = -3.0e38f;
    for (int i = 0; i < 32; ++i) {
      int j = lane + i * 32;
      float x = (s_sc[r * MM + j] >= thr) ? s_lg[r * MM + j] : -3.0e38f;
      s_lg[r * MM + j] = x;
      mx = fmaxf(mx, x);
    }
    mx = wave_max32(mx);
    float sum = 0.0f;
    for (int i = 0; i < 32; ++i) {
      int j = lane + i * 32;
      float x = s_lg[r * MM + j];
      float p = (x > -1.0e37f) ? __expf(x - mx) : 0.0f;
      s_lg[r * MM + j] = p;
      sum += p;
    }
    sum = wave_sum32(sum);
    float inv = 1.0f / fmaxf(sum, 1.0e-30f);
    for (int i = 0; i < 32; ++i) s_lg[r * MM + lane + i * 32] *= inv;
  }
  __syncthreads();

  // ---- phase 3b: attn -> bf16 (reuses scores region) ----
  bf16* s_attn = (bf16*)s_sc;
  for (int rr = 0; rr < 2; ++rr) {
    int r = wave * 2 + rr;
    for (int i = 0; i < 32; ++i) {
      int j = lane + i * 32;
      s_attn[r * MM + j] = (bf16)s_lg[r * MM + j];
    }
  }
  __syncthreads();

  // ---- phase 4: out_tile = attn @ em_V (emVt is feature-major bf16) ----
  bf16* s_outbf = (bf16*)s_lg;
  {
    int n0 = wave * 16;
    const bf16* arow = s_attn + (lane & 15) * MM;
    const bf16* vcol = emVt_bf + ((size_t)b * DD + n0 + (lane & 15)) * MM;
    v8f acc = {};
    for (int k0 = 0; k0 < MM; k0 += 32)
      acc = wmma_bf(frag_a(arow, k0, lane), frag_b(vcol, k0, lane), acc);
    int col = n0 + (lane & 15);
#pragma unroll
    for (int v = 0; v < 8; ++v) s_outbf[(mh + v) * DD + col] = (bf16)acc[v];
  }
  __syncthreads();

  // ---- phase 5: out = out_tile @ Wo[b%4] + bo ----
  {
    int e = wave * 16 + (lane & 15);
    const bf16* arow = s_outbf + (lane & 15) * DD;
    const bf16* wcol = Wot_bf + ((size_t)cblk * DD + e) * DD;
    v8f acc = {};
#pragma unroll
    for (int kk = 0; kk < 4; ++kk)
      acc = wmma_bf(frag_a(arow, kk * 32, lane), frag_b(wcol, kk * 32, lane), acc);
    float bias = bo[cblk * DD + e];
#pragma unroll
    for (int v = 0; v < 8; ++v) {
      int r = mh + v;
      out[((size_t)b * NCC + row0 + r) * DD + e] = acc[v] + bias;
    }
  }
}

// ---------------------------------------------------------------------------
// K3: top-64 candidates per batch, candidate normalization.
__global__ __launch_bounds__(256) void cand_kernel(
    const float* __restrict__ novelty, const float* __restrict__ q_nov,
    const float* __restrict__ v_nov, float* __restrict__ cand_w,
    float* __restrict__ candKn, bf16* __restrict__ candKn_bf,
    float* __restrict__ candV) {
  __shared__ float s_nov[NCC];
  __shared__ float s_rv[256];
  __shared__ int   s_ri[256];
  __shared__ float s_val[CC2];
  __shared__ int   s_idx[CC2];
  __shared__ float s_sum;
  int b = blockIdx.x;
  int tid = threadIdx.x;
  int lane = tid & 31, wave = tid >> 5;
  for (int i = tid; i < NCC; i += 256) s_nov[i] = novelty[(size_t)b * NCC + i];
  __syncthreads();
  for (int k = 0; k < CC2; ++k) {
    float m = -3.0e38f; int mi = 0;
    for (int i = 0; i < 8; ++i) {
      int j = tid + i * 256;
      float x = s_nov[j];
      if (x > m) { m = x; mi = j; }
    }
    s_rv[tid] = m; s_ri[tid] = mi;
    __syncthreads();
    if (tid == 0) {
      float bm = -3.0e38f; int bi = 0;
      for (int t2 = 0; t2 < 256; ++t2)
        if (s_rv[t2] > bm) { bm = s_rv[t2]; bi = s_ri[t2]; }
      s_val[k] = bm; s_idx[k] = bi; s_nov[bi] = -3.0e38f;
    }
    __syncthreads();
  }
  if (tid == 0) {
    float s = 0.0f;
    for (int k = 0; k < CC2; ++k) s += s_val[k];
    s_sum = s;
  }
  __syncthreads();
  if (tid < CC2) cand_w[b * CC2 + tid] = s_val[tid] / (s_sum + 1e-8f);
  for (int c = wave; c < CC2; c += 8) {
    int idx = s_idx[c];
    const float* src = q_nov + ((size_t)b * NCC + idx) * DD + lane * 4;
    float v0 = src[0], v1 = src[1], v2 = src[2], v3 = src[3];
    float ss = wave_sum32(v0 * v0 + v1 * v1 + v2 * v2 + v3 * v3);
    float inv = 1.0f / fmaxf(sqrtf(ss), 1e-8f);
    size_t base = ((size_t)b * CC2 + c) * DD + lane * 4;
    float k0 = v0 * inv, k1 = v1 * inv, k2 = v2 * inv, k3 = v3 * inv;
    candKn[base + 0] = k0; candKn[base + 1] = k1;
    candKn[base + 2] = k2; candKn[base + 3] = k3;
    candKn_bf[base + 0] = (bf16)k0; candKn_bf[base + 1] = (bf16)k1;
    candKn_bf[base + 2] = (bf16)k2; candKn_bf[base + 3] = (bf16)k3;
    const float* vs = v_nov + ((size_t)b * NCC + idx) * DD + lane * 4;
    candV[base + 0] = vs[0]; candV[base + 1] = vs[1];
    candV[base + 2] = vs[2]; candV[base + 3] = vs[3];
  }
}

// ---------------------------------------------------------------------------
// K4: slot_scores = candKn @ em_K^T - 0.5*em_S; slot softmax; alpha.
__global__ __launch_bounds__(256) void slot_kernel(
    const float* __restrict__ em_S, const float* __restrict__ tau,
    const float* __restrict__ g_em, const float* __restrict__ cand_w,
    const bf16* __restrict__ candKn_bf, const bf16* __restrict__ emK_bf,
    float* __restrict__ alpha) {
  extern __shared__ __align__(16) unsigned char smem[];
  float* s_ss = (float*)smem;  // 16*1024
  int b = blockIdx.y, ct = blockIdx.x;
  int lane = threadIdx.x & 31, wave = threadIdx.x >> 5;
  int mh = (lane & 16) ? 8 : 0;
  const bf16* arow = candKn_bf + ((size_t)b * CC2 + ct * 16 + (lane & 15)) * DD;
  v16bf a[4];
#pragma unroll
  for (int kk = 0; kk < 4; ++kk) a[kk] = frag_a(arow, kk * 32, lane);
  for (int t = wave; t < MM / 16; t += 8) {
    int slot = t * 16 + (lane & 15);
    const bf16* kcol = emK_bf + ((size_t)b * MM + slot) * DD;
    v8f acc = {};
#pragma unroll
    for (int kk = 0; kk < 4; ++kk)
      acc = wmma_bf(a[kk], frag_b(kcol, kk * 32, lane), acc);
    float hs = 0.5f * em_S[(size_t)b * MM + slot];
#pragma unroll
    for (int v = 0; v < 8; ++v) s_ss[(mh + v) * MM + slot] = acc[v] - hs;
  }
  __syncthreads();
  float invtau = 1.0f / fmaxf(tau[b], 0.01f);
  for (int rr = 0; rr < 2; ++rr) {
    int r = wave * 2 + rr;
    int cand = ct * 16 + r;
    float mx = -3.0e38f;
    for (int i = 0; i < 32; ++i) {
      int j = lane + i * 32;
      float x = s_ss[r * MM + j] * invtau;
      s_ss[r * MM + j] = x;
      mx = fmaxf(mx, x);
    }
    mx = wave_max32(mx);
    float sum = 0.0f;
    for (int i = 0; i < 32; ++i) {
      int j = lane + i * 32;
      float p = __expf(s_ss[r * MM + j] - mx);
      s_ss[r * MM + j] = p;
      sum += p;
    }
    sum = wave_sum32(sum);
    float coef = g_em[b] * cand_w[b * CC2 + cand] / sum;
    for (int i = 0; i < 32; ++i) {
      int j = lane + i * 32;
      alpha[((size_t)b * CC2 + cand) * MM + j] = s_ss[r * MM + j] * coef;
    }
  }
}

// ---------------------------------------------------------------------------
// K5: EMA memory update, one slot per wave (f32 exact path).
__global__ __launch_bounds__(256) void update_kernel(
    const float* __restrict__ alpha, const float* __restrict__ candKn,
    const float* __restrict__ candV, const float* __restrict__ em_K,
    const float* __restrict__ em_V, const float* __restrict__ em_S,
    const float* __restrict__ em_age, const float* __restrict__ decay,
    float* __restrict__ outK, float* __restrict__ outV,
    float* __restrict__ outS, float* __restrict__ outAge) {
  int lane = threadIdx.x & 31, wave = threadIdx.x >> 5;
  int gs = blockIdx.x * 8 + wave;
  int b = gs >> 10, m = gs & (MM - 1);
  const float* ab = alpha + (size_t)b * CC2 * MM + m;
  float asum = ab[(size_t)lane * MM] + ab[(size_t)(lane + 32) * MM];
  asum = wave_sum32(asum);
  float accK[4] = {0, 0, 0, 0}, accV[4] = {0, 0, 0, 0};
  for (int c = 0; c < CC2; ++c) {
    float al = ab[(size_t)c * MM];
    const float* kr = candKn + ((size_t)b * CC2 + c) * DD + lane * 4;
    const float* vr = candV  + ((size_t)b * CC2 + c) * DD + lane * 4;
#pragma unroll
    for (int j = 0; j < 4; ++j) { accK[j] += al * kr[j]; accV[j] += al * vr[j]; }
  }
  float denom = fmaxf(asum, 1e-8f);
  float bK[4], bV[4], ss = 0.0f;
#pragma unroll
  for (int j = 0; j < 4; ++j) {
    bK[j] = accK[j] / denom; bV[j] = accV[j] / denom; ss += bK[j] * bK[j];
  }
  ss = wave_sum32(ss);
  float invn = 1.0f / fmaxf(sqrtf(ss), 1e-8f);
  bool upd = asum > 1e-8f;
  float a = fminf(asum, 1.0f);
  size_t base = ((size_t)b * MM + m) * DD + lane * 4;
#pragma unroll
  for (int j = 0; j < 4; ++j) {
    float ek = em_K[base + j], ev = em_V[base + j];
    outK[base + j] = upd ? (1.0f - a) * ek + a * (bK[j] * invn) : ek;
    outV[base + j] = upd ? (1.0f - a) * ev + a * bV[j] : ev;
  }
  if (lane == 0) {
    float s = fminf(fmaxf(em_S[(size_t)b * MM + m] + asum, 0.0f), 4.0f) * decay[b];
    outS[(size_t)b * MM + m] = s;
    outAge[(size_t)b * MM + m] = em_age[(size_t)b * MM + m] * (1.0f - asum);
  }
}

// K6a: deterministic per-batch strength totals.
__global__ __launch_bounds__(256) void totals_kernel(float* __restrict__ totals,
                                                     const float* __restrict__ outS) {
  __shared__ float red[256];
  int b = blockIdx.x, tid = threadIdx.x;
  float s = 0.0f;
  for (int i = tid; i < MM; i += 256) s += outS[(size_t)b * MM + i];
  red[tid] = s;
  __syncthreads();
  for (int off = 128; off > 0; off >>= 1) {
    if (tid < off) red[tid] += red[tid + off];
    __syncthreads();
  }
  if (tid == 0) totals[b] = red[0];
}

// K6b: budget enforcement.
__global__ void budget_kernel(float* __restrict__ outS, const float* __restrict__ totals) {
  int i = blockIdx.x * blockDim.x + threadIdx.x;
  if (i < BSB * MM) {
    int b = i >> 10;
    outS[i] *= fminf(1.0f, 64.0f / fmaxf(totals[b], 1e-8f));
  }
}

// ---------------------------------------------------------------------------
extern "C" void kernel_launch(void* const* d_in, const int* in_sizes, int n_in,
                              void* d_out, int out_size, void* d_ws, size_t ws_size,
                              hipStream_t stream) {
  const float* q        = (const float*)d_in[0];
  const float* q_nov    = (const float*)d_in[1];
  const float* v_nov    = (const float*)d_in[2];
  const float* surprise = (const float*)d_in[3];
  const float* w_nov    = (const float*)d_in[4];
  const float* g_em     = (const float*)d_in[5];
  const float* tau      = (const float*)d_in[6];
  const float* decay    = (const float*)d_in[7];
  const float* em_K     = (const float*)d_in[8];
  const float* em_V     = (const float*)d_in[9];
  const float* em_S     = (const float*)d_in[10];
  const float* em_age   = (const float*)d_in[11];
  const float* Wq       = (const float*)d_in[12];
  const float* bq       = (const float*)d_in[13];
  const float* Wo       = (const float*)d_in[14];
  const float* bo       = (const float*)d_in[15];

  float* out    = (float*)d_out;
  float* outK   = out  + (size_t)BSB * NCC * DD;
  float* outV   = outK + (size_t)BSB * MM * DD;
  float* outS   = outV + (size_t)BSB * MM * DD;
  float* outAge = outS + (size_t)BSB * MM;

  char* w = (char*)d_ws;
  bf16*  q_bf      = (bf16*)w;  w += (size_t)BSB * NCC * DD * 2;
  bf16*  qsum_bf   = (bf16*)w;  w += (size_t)BSB * NCC * DD * 2;
  bf16*  qnov_bf   = (bf16*)w;  w += (size_t)BSB * NCC * DD * 2;
  bf16*  emK_bf    = (bf16*)w;  w += (size_t)BSB * MM * DD * 2;
  bf16*  emVt_bf   = (bf16*)w;  w += (size_t)BSB * MM * DD * 2;
  bf16*  Wqt_bf    = (bf16*)w;  w += (size_t)4 * DD * DD * 2;
  bf16*  Wot_bf    = (bf16*)w;  w += (size_t)4 * DD * DD * 2;
  float* novelty   = (float*)w; w += (size_t)BSB * NCC * 4;
  float* cand_w    = (float*)w; w += (size_t)BSB * CC2 * 4;
  float* candKn    = (float*)w; w += (size_t)BSB * CC2 * DD * 4;
  bf16*  candKn_bf = (bf16*)w;  w += (size_t)BSB * CC2 * DD * 2;
  float* candV     = (float*)w; w += (size_t)BSB * CC2 * DD * 4;
  float* alpha     = (float*)w; w += (size_t)BSB * CC2 * MM * 4;
  float* totals    = (float*)w; w += 64;

  int nq = BSB * NCC * DD;
  int nk = BSB * MM * DD;
  cvt_bf16_kernel<<<(nq + 255) / 256, 256, 0, stream>>>(q_bf, q, nq);
  cvt_bf16_kernel<<<(nq + 255) / 256, 256, 0, stream>>>(qnov_bf, q_nov, nq);
  cvt_bf16_kernel<<<(nk + 255) / 256, 256, 0, stream>>>(emK_bf, em_K, nk);
  tcvt_kernel<<<dim3((MM * DD + 255) / 256, BSB), 256, 0, stream>>>(emVt_bf, em_V, MM, DD);
  tcvt_kernel<<<dim3((DD * DD + 255) / 256, 4), 256, 0, stream>>>(Wqt_bf, Wq, DD, DD);
  tcvt_kernel<<<dim3((DD * DD + 255) / 256, 4), 256, 0, stream>>>(Wot_bf, Wo, DD, DD);

  qcross_kernel<<<dim3(NCC / 16, BSB), 128, 0, stream>>>(q, bq, q_bf, Wqt_bf, qsum_bf);

  size_t attn_smem = (size_t)(16 * MM * 2 + 16 + 128) * sizeof(float);
  attn_kernel<<<dim3(NCC / 16, BSB), 256, attn_smem, stream>>>(
      em_S, surprise, w_nov, bo, q_bf, qsum_bf, qnov_bf, emK_bf, emVt_bf, Wot_bf,
      out, novelty);

  cand_kernel<<<BSB, 256, 0, stream>>>(novelty, q_nov, v_nov, cand_w, candKn,
                                       candKn_bf, candV);

  size_t slot_smem = (size_t)16 * MM * sizeof(float);
  slot_kernel<<<dim3(4, BSB), 256, slot_smem, stream>>>(em_S, tau, g_em, cand_w,
                                                        candKn_bf, emK_bf, alpha);

  update_kernel<<<(BSB * MM) / 8, 256, 0, stream>>>(alpha, candKn, candV, em_K, em_V,
                                                    em_S, em_age, decay, outK, outV,
                                                    outS, outAge);
  totals_kernel<<<BSB, 256, 0, stream>>>(totals, outS);
  budget_kernel<<<(BSB * MM + 255) / 256, 256, 0, stream>>>(outS, totals);

  (void)in_sizes; (void)n_in; (void)out_size; (void)ws_size;
}
